// ALayer_DR1_v1_light_14637248545086
// MI455X (gfx1250) — matrix operand order repaired and checked by reading, not compiled, not run on hardware
//
#include <hip/hip_runtime.h>
#include <math.h>

typedef __attribute__((ext_vector_type(16))) __bf16 v16bf;
typedef __attribute__((ext_vector_type(8)))  __bf16 v8bf;
typedef __attribute__((ext_vector_type(8)))  float  v8f;
typedef __attribute__((ext_vector_type(4)))  float  v4f;
typedef int v4i_vs __attribute__((vector_size(16)));   // matches builtin param type

#define HWDIM 56
#define HWTOT 3136      // 56*56
#define CDIM  256
#define BDIM  8
#define SASTR 264       // 256 + 8 pad -> 528B row stride, conflict-free b128 reads

union AFrag { v16bf v; v8bf h[2]; unsigned u[8]; };
union BFrag { v16bf v; unsigned u[8]; };

// Async global->LDS path (CDNA5): guarded so the file compiles on toolchains
// where the builtin is absent; fallback is VGPR round-trip + ds_store_b128.
#if defined(__has_builtin)
#if __has_builtin(__builtin_amdgcn_global_load_async_to_lds_b128) && \
    __has_builtin(__builtin_amdgcn_s_wait_asynccnt)
#define USE_ASYNC_LDS 1
#endif
#endif
#ifndef USE_ASYNC_LDS
#define USE_ASYNC_LDS 0
#endif

// ---------------------------------------------------------------------------
// 1) x [B,C,H,W] fp32  ->  xb [B,H,W,C] bf16   (NHWC so K=channel is contiguous)
// ---------------------------------------------------------------------------
__global__ void k_x_to_nhwc_bf16(const float* __restrict__ x,
                                 __bf16* __restrict__ xb, int total) {
  int g = blockIdx.x * blockDim.x + threadIdx.x;
  if (g >= total) return;
  int c    = g & (CDIM - 1);
  int rest = g >> 8;
  int p    = rest % HWTOT;
  int b    = rest / HWTOT;
  xb[g] = (__bf16)x[((size_t)(b * CDIM + c)) * HWTOT + p];
}

// ---------------------------------------------------------------------------
// 2) pack modulated weights (weight * A_w) into WMMA B-fragment lane layout.
// Fragment f = (tap*16 + ocTile)*8 + kTile ; 512 bf16 per fragment.
// Lane L holds N = L%16, K = 16*(L/16) + 2v + {0,1} (v = 0..7), contiguous 32B.
// ---------------------------------------------------------------------------
__global__ void k_pack_w(const float* __restrict__ weight,
                         const float* __restrict__ aw,
                         __bf16* __restrict__ pb) {
  int tid  = blockIdx.x * blockDim.x + threadIdx.x;     // 36864 threads
  int lane = tid & 31;
  int kt   = (tid >> 5) & 7;
  int ocT  = (tid >> 8) & 15;
  int t    = tid >> 12;
  if (t >= 9) return;
  int oc   = ocT * 16 + (lane & 15);
  int half = lane >> 4;
  size_t fragBase = ((size_t)((t * 16 + ocT) * 8 + kt)) * 512;
  __bf16* dst = pb + fragBase + (size_t)lane * 16;
#pragma unroll
  for (int v = 0; v < 8; ++v) {
    int c0 = kt * 32 + 16 * half + 2 * v;
    float w0 = weight[((size_t)oc * CDIM + c0)     * 9 + t] * aw[(c0)     * 9 + t];
    float w1 = weight[((size_t)oc * CDIM + c0 + 1) * 9 + t] * aw[(c0 + 1) * 9 + t];
    dst[2 * v]     = (__bf16)w0;
    dst[2 * v + 1] = (__bf16)w1;
  }
}

// ---------------------------------------------------------------------------
// 3) SE conv1 (256 -> 16) + relu.  One thread per (b,pixel); se_w1 staged
//    through LDS in 64-channel chunks.  r1 layout: [b][p][j0..15].
// ---------------------------------------------------------------------------
__global__ void k_se1(const float* __restrict__ x,
                      const float* __restrict__ w1,
                      float* __restrict__ r1) {
  __shared__ float sw[64 * 9 * 16];                     // 36 KB
  int g = blockIdx.x * blockDim.x + threadIdx.x;        // 98*256 = 25088 exact
  int b = g / HWTOT;
  int p = g % HWTOT;
  int h = p / HWDIM, w = p % HWDIM;
  float acc[16];
#pragma unroll
  for (int j = 0; j < 16; ++j) acc[j] = 0.f;

  for (int cc = 0; cc < CDIM; cc += 64) {
    __syncthreads();
    for (int i = threadIdx.x; i < 9216; i += 256) {
      int j  = i & 15;
      int t  = (i >> 4) % 9;
      int cl = i / 144;
      sw[i] = w1[((size_t)j * CDIM + (cc + cl)) * 9 + t];
    }
    __syncthreads();
    for (int t = 0; t < 9; ++t) {
      int hh = h + t / 3 - 1, ww = w + t % 3 - 1;
      bool valid = ((unsigned)hh < (unsigned)HWDIM) & ((unsigned)ww < (unsigned)HWDIM);
      int pn = valid ? (hh * HWDIM + ww) : 0;
      const float* xp = x + (size_t)b * CDIM * HWTOT + pn;
      for (int cl = 0; cl < 64; ++cl) {
        float xv = valid ? xp[(size_t)(cc + cl) * HWTOT] : 0.f;
        const v4f* swv = (const v4f*)&sw[(cl * 9 + t) * 16];
#pragma unroll
        for (int j4 = 0; j4 < 4; ++j4) {
          v4f wv = swv[j4];
          acc[j4 * 4 + 0] += xv * wv[0];
          acc[j4 * 4 + 1] += xv * wv[1];
          acc[j4 * 4 + 2] += xv * wv[2];
          acc[j4 * 4 + 3] += xv * wv[3];
        }
      }
    }
  }
  v4f* o = (v4f*)&r1[(size_t)g * 16];
#pragma unroll
  for (int j4 = 0; j4 < 4; ++j4) {
    v4f v;
#pragma unroll
    for (int e = 0; e < 4; ++e) v[e] = fmaxf(acc[j4 * 4 + e], 0.f);
    o[j4] = v;
  }
}

// ---------------------------------------------------------------------------
// 4) SE conv2 (16 -> 1) + sigmoid -> attention map a[b][p]
// ---------------------------------------------------------------------------
__global__ void k_se2(const float* __restrict__ r1,
                      const float* __restrict__ w2,
                      float* __restrict__ amap) {
  int g = blockIdx.x * blockDim.x + threadIdx.x;
  if (g >= BDIM * HWTOT) return;
  int b = g / HWTOT, p = g % HWTOT;
  int h = p / HWDIM, w = p % HWDIM;
  float acc = 0.f;
  for (int t = 0; t < 9; ++t) {
    int hh = h + t / 3 - 1, ww = w + t % 3 - 1;
    if ((unsigned)hh >= (unsigned)HWDIM || (unsigned)ww >= (unsigned)HWDIM) continue;
    const float* rrow = &r1[((size_t)b * HWTOT + hh * HWDIM + ww) * 16];
#pragma unroll
    for (int j = 0; j < 16; ++j) acc += rrow[j] * w2[j * 9 + t];
  }
  amap[g] = 1.f / (1.f + __expf(-acc));
}

// ---------------------------------------------------------------------------
// Stage one tap's shifted 32-pixel x 256-channel bf16 tile into LDS.
// Thread i handles pixel i/8, 32-channel chunk (i%8)*32 (64 bytes).
// Invalid (padding) pixels are zero-filled with ds_store_b128.
// ---------------------------------------------------------------------------
__device__ __forceinline__ void stage_tap(const __bf16* __restrict__ xbase,
                                          __bf16* dstBuf,
                                          int pix0, int t, int tix) {
  const int pi = tix >> 3;
  const int ci = (tix & 7) * 32;
  const int p  = pix0 + pi;
  const int h  = p / HWDIM, w = p % HWDIM;
  const int hh = h + t / 3 - 1, ww = w + t % 3 - 1;
  const bool valid = ((unsigned)hh < (unsigned)HWDIM) & ((unsigned)ww < (unsigned)HWDIM);
  __bf16* dst = dstBuf + pi * SASTR + ci;
  if (valid) {
    const __bf16* src = xbase + ((size_t)(hh * HWDIM + ww)) * CDIM + ci;
#if USE_ASYNC_LDS
#pragma unroll
    for (int q = 0; q < 4; ++q) {
      __builtin_amdgcn_global_load_async_to_lds_b128(
          (__attribute__((address_space(1))) v4i_vs*)(uintptr_t)(src + q * 8),
          (__attribute__((address_space(3))) v4i_vs*)(unsigned)(uintptr_t)(dst + q * 8),
          0, 0);
    }
#else
    const v8bf* sv = (const v8bf*)src;
    v8bf d0 = sv[0], d1 = sv[1], d2 = sv[2], d3 = sv[3];
    v8bf* dv = (v8bf*)dst;
    dv[0] = d0; dv[1] = d1; dv[2] = d2; dv[3] = d3;
#endif
  } else {
    v8bf z = {};
    v8bf* dv = (v8bf*)dst;
    dv[0] = z; dv[1] = z; dv[2] = z; dv[3] = z;
  }
}

// ---------------------------------------------------------------------------
// 5) Main conv as 9 shifted GEMMs with V_WMMA_F32_16X16X32_BF16.
//    A tile double-buffered in LDS (shared by all 8 waves); async staged.
//    Wave: two 16-pixel x 16-oc tiles (shared B fragment). 144 WMMA / wave.
//    Epilogue fuses out = acc * a[b,p], coalesced b128 stores (NCHW).
// ---------------------------------------------------------------------------
__global__ void __launch_bounds__(256)
k_conv_wmma(const __bf16* __restrict__ xb, const __bf16* __restrict__ pb,
            const float* __restrict__ amap, float* __restrict__ out) {
  __shared__ __bf16 sA[2][32 * SASTR];                  // 2 x 16.5 KB

  const int tix  = threadIdx.x;
  const int lane = tix & 31;
  const int wave = tix >> 5;
  const int b    = blockIdx.z;
  const int ocT  = blockIdx.y * 8 + wave;   // 0..15
  const int pix0 = blockIdx.x * 32;         // two 16-pixel tiles
  const int half = lane >> 4;
  const int m    = lane & 15;

  const __bf16* xbase = xb + (size_t)b * HWTOT * CDIM;

  v8f acc0 = {};
  v8f acc1 = {};

  // prologue: stage tap 0
  stage_tap(xbase, sA[0], pix0, 0, tix);
#if USE_ASYNC_LDS
  __builtin_amdgcn_s_wait_asynccnt(0);
#endif
  __syncthreads();

  for (int t = 0; t < 9; ++t) {
    const __bf16* cur = sA[t & 1];
    if (t < 8) stage_tap(xbase, sA[(t + 1) & 1], pix0, t + 1, tix);

    const __bf16* bfp = pb + ((size_t)((t * 16 + ocT) * 8)) * 512 + (size_t)lane * 16;
    if (t < 8) __builtin_prefetch(bfp + 16 * 8 * 512, 0, 3);   // next tap's B frags

    const int base0 = m * SASTR;
    const int base1 = (16 + m) * SASTR;
#pragma unroll
    for (int kt = 0; kt < 8; ++kt) {
      BFrag bf;
      bf.v = *(const v16bf*)(bfp + (size_t)kt * 512);

      const int ko = kt * 32 + 16 * half;
      AFrag a0, a1;
      a0.h[0] = *(const v8bf*)&cur[base0 + ko];
      a0.h[1] = *(const v8bf*)&cur[base0 + ko + 8];
      a1.h[0] = *(const v8bf*)&cur[base1 + ko];
      a1.h[1] = *(const v8bf*)&cur[base1 + ko + 8];

      acc0 = __builtin_amdgcn_wmma_f32_16x16x32_bf16(false, a0.v, false, bf.v,
                                                     (short)0, acc0, false, false);
      acc1 = __builtin_amdgcn_wmma_f32_16x16x32_bf16(false, a1.v, false, bf.v,
                                                     (short)0, acc1, false, false);
    }

#if USE_ASYNC_LDS
    __builtin_amdgcn_s_wait_asynccnt(0);     // next buffer fully staged
#endif
    __syncthreads();                          // all waves done with `cur`
  }

  // Epilogue: D tile layout -> lane has N=m (oc), rows M=r+8*half (pixels).
  const int oc    = ocT * 16 + m;
  const int prow0 = pix0 + 8 * half;
  const int prow1 = pix0 + 16 + 8 * half;
  const float* am0 = amap + (size_t)b * HWTOT + prow0;
  const float* am1 = amap + (size_t)b * HWTOT + prow1;
  float* o0 = out + ((size_t)(b * CDIM + oc)) * HWTOT + prow0;
  float* o1 = out + ((size_t)(b * CDIM + oc)) * HWTOT + prow1;

  v4f a0lo = *(const v4f*)am0, a0hi = *(const v4f*)(am0 + 4);
  v4f a1lo = *(const v4f*)am1, a1hi = *(const v4f*)(am1 + 4);

  union { v8f v; v4f q[2]; } s0, s1;
#pragma unroll
  for (int r = 0; r < 4; ++r) {
    s0.v[r]     = acc0[r]     * a0lo[r];
    s0.v[r + 4] = acc0[r + 4] * a0hi[r];
    s1.v[r]     = acc1[r]     * a1lo[r];
    s1.v[r + 4] = acc1[r + 4] * a1hi[r];
  }
  *(v4f*)o0       = s0.q[0];
  *(v4f*)(o0 + 4) = s0.q[1];
  *(v4f*)o1       = s1.q[0];
  *(v4f*)(o1 + 4) = s1.q[1];
}

// ---------------------------------------------------------------------------
extern "C" void kernel_launch(void* const* d_in, const int* in_sizes, int n_in,
                              void* d_out, int out_size, void* d_ws, size_t ws_size,
                              hipStream_t stream) {
  const float* x      = (const float*)d_in[0];
  const float* weight = (const float*)d_in[1];
  const float* A_w    = (const float*)d_in[2];
  const float* se_w1  = (const float*)d_in[3];
  const float* se_w2  = (const float*)d_in[4];
  float* out = (float*)d_out;

  char* ws = (char*)d_ws;
  __bf16* xb   = (__bf16*)(ws);                    // 12,845,056 B
  __bf16* pb   = (__bf16*)(ws + 12845056);         //  1,179,648 B
  float*  r1   = (float*)(ws + 14024704);          //  1,605,632 B
  float*  amap = (float*)(ws + 15630336);          //    100,352 B
  (void)in_sizes; (void)n_in; (void)out_size; (void)ws_size;

  const int totalX = BDIM * HWTOT * CDIM;          // 6,422,528
  k_x_to_nhwc_bf16<<<(totalX + 255) / 256, 256, 0, stream>>>(x, xb, totalX);
  k_pack_w<<<144, 256, 0, stream>>>(weight, A_w, pb);
  k_se1<<<98, 256, 0, stream>>>(x, se_w1, r1);
  k_se2<<<(BDIM * HWTOT + 255) / 256, 256, 0, stream>>>(r1, se_w2, amap);

  dim3 grid(98, 2, BDIM);                          // 98 pixel-pairs, 2x8 ocTiles, 8 batch
  k_conv_wmma<<<grid, 256, 0, stream>>>(xb, pb, amap, out);
}
